// SlidingWindowAttention_33844342292528
// MI455X (gfx1250) — compile-verified, matrix operand order checked by reading, not run on hardware
//
#include <hip/hip_runtime.h>
#include <hip/hip_bf16.h>

// ---------------------------------------------------------------------------
// Types for CDNA5 WMMA (gfx1250, wave32)
// ---------------------------------------------------------------------------
typedef __attribute__((ext_vector_type(16))) __bf16 v16bf;
typedef __attribute__((ext_vector_type(8)))  __bf16 v8bf;
typedef __attribute__((ext_vector_type(4)))  __bf16 v4bf;
typedef __attribute__((ext_vector_type(8)))  float  v8f;

#define B_    2
#define S_    2048
#define DIM_  2048
#define H_    16
#define HD_   128
#define WIN_  512
#define MTOT  (B_ * S_)          // 4096 rows

union V16u { v16bf v; v8bf h[2]; };

// Per-lane WMMA 16-bit A/B operand fetch from a K-contiguous row:
// lane holds row/col = lane&15; K halves [8*(lane>=16), +8) and [16+8*(lane>=16), +8).
__device__ __forceinline__ V16u load_op16(const __bf16* p) {
    V16u r;
    r.h[0] = *(const v8bf*)(p);
    r.h[1] = *(const v8bf*)(p + 16);
    return r;
}

__device__ __forceinline__ v8f wmma_bf16(v16bf a, v16bf b, v8f c) {
    // (neg_a, A, neg_b, B, c_mod, C, reuse_a, reuse_b)
    return __builtin_amdgcn_wmma_f32_16x16x32_bf16(false, a, false, b, (short)0, c,
                                                   false, false);
}

// ---------------------------------------------------------------------------
// Kernel 1: f32 -> bf16 staging (4 elems / thread)
// ---------------------------------------------------------------------------
__global__ __launch_bounds__(256) void cvt_f32_bf16_v4(const float* __restrict__ src,
                                                       __bf16* __restrict__ dst,
                                                       int n4) {
    int i = blockIdx.x * 256 + threadIdx.x;
    if (i >= n4) return;
    float4 f = ((const float4*)src)[i];
    v4bf o;
    o.x = (__bf16)f.x; o.y = (__bf16)f.y; o.z = (__bf16)f.z; o.w = (__bf16)f.w;
    ((v4bf*)dst)[i] = o;
}

// ---------------------------------------------------------------------------
// Kernel 2: NT GEMM  O[m,n] = sum_k A[m,k] * W[n,k]   (bf16 in, f32 out)
// Block = 256 thr = 8 waves, block tile 128x128, wave tile 32x64.
// mode 0: scatter output to [b,h,s,hd] (q/k/v);  mode 1: plain [m,n].
// blockIdx.z selects (W0,O0)/(W1,O1)/(W2,O2).
// ---------------------------------------------------------------------------
__global__ __launch_bounds__(256) void gemm_nt_bf16(
    const __bf16* __restrict__ A,
    const __bf16* __restrict__ W0, const __bf16* __restrict__ W1,
    const __bf16* __restrict__ W2,
    float* __restrict__ O0, float* __restrict__ O1, float* __restrict__ O2,
    int K, int N, int mode) {
    const int lane  = threadIdx.x & 31;
    const int wave  = threadIdx.x >> 5;
    const int waveM = wave & 3;     // 4 waves along M
    const int waveN = wave >> 2;    // 2 waves along N
    const int m0 = blockIdx.y * 128 + waveM * 32;
    const int n0 = blockIdx.x * 128 + waveN * 64;
    const int lrow  = lane & 15;
    const int lhalf = (lane >> 4) << 3;   // 0 or 8

    const __bf16* W = (blockIdx.z == 0) ? W0 : (blockIdx.z == 1) ? W1 : W2;
    float*        O = (blockIdx.z == 0) ? O0 : (blockIdx.z == 1) ? O1 : O2;

    v8f acc[2][4];
#pragma unroll
    for (int mt = 0; mt < 2; ++mt)
#pragma unroll
        for (int nt = 0; nt < 4; ++nt) acc[mt][nt] = 0.0f;

    for (int k = 0; k < K; k += 32) {
        V16u a[2];
#pragma unroll
        for (int mt = 0; mt < 2; ++mt)
            a[mt] = load_op16(A + (size_t)(m0 + mt * 16 + lrow) * K + k + lhalf);
        V16u bm[4];
#pragma unroll
        for (int nt = 0; nt < 4; ++nt)
            bm[nt] = load_op16(W + (size_t)(n0 + nt * 16 + lrow) * K + k + lhalf);
#pragma unroll
        for (int mt = 0; mt < 2; ++mt)
#pragma unroll
            for (int nt = 0; nt < 4; ++nt)
                acc[mt][nt] = wmma_bf16(a[mt].v, bm[nt].v, acc[mt][nt]);
    }

    // C/D layout: VGPR r -> row r + 8*(lane>=16), lanes -> col lane&15
#pragma unroll
    for (int mt = 0; mt < 2; ++mt)
#pragma unroll
        for (int nt = 0; nt < 4; ++nt)
#pragma unroll
            for (int r = 0; r < 8; ++r) {
                int m = m0 + mt * 16 + r + lhalf;
                int n = n0 + nt * 16 + lrow;
                float v = acc[mt][nt][r];
                if (mode == 0) {
                    int b = m >> 11, s = m & (S_ - 1);
                    int h = n >> 7,  hd = n & (HD_ - 1);
                    O[(((size_t)(b * H_ + h)) * S_ + s) * HD_ + hd] = v;
                } else {
                    O[(size_t)m * N + n] = v;
                }
            }
}

// ---------------------------------------------------------------------------
// Kernel 3: RoPE on q,k (f32 -> bf16, [b,h,s,hd]); V f32 -> bf16 transposed
// to [b,h,hd,s]. One thread per (b,h,s,pair t).
// ---------------------------------------------------------------------------
__global__ __launch_bounds__(256) void rope_cvt(
    const float* __restrict__ qf, const float* __restrict__ kf,
    const float* __restrict__ vf,
    __bf16* __restrict__ qb, __bf16* __restrict__ kb, __bf16* __restrict__ vt) {
    int idx = blockIdx.x * 256 + threadIdx.x;      // B*H*S*64
    int t  = idx & 63;
    int s  = (idx >> 6) & (S_ - 1);
    int bh = idx >> 17;                            // 0..31
    size_t base = ((size_t)bh * S_ + s) * HD_;

    // inv_freq = 10000^(-2t/HD)
    float inv = __expf(-((float)(2 * t) * (1.0f / (float)HD_)) * 9.210340371976184f);
    float ang = (float)s * inv;
    float sn, c;
    __sincosf(ang, &sn, &c);

    float x1 = qf[base + 2 * t], x2 = qf[base + 2 * t + 1];
    qb[base + 2 * t]     = (__bf16)(x1 * c - x2 * sn);
    qb[base + 2 * t + 1] = (__bf16)(x1 * sn + x2 * c);

    x1 = kf[base + 2 * t]; x2 = kf[base + 2 * t + 1];
    kb[base + 2 * t]     = (__bf16)(x1 * c - x2 * sn);
    kb[base + 2 * t + 1] = (__bf16)(x1 * sn + x2 * c);

    size_t vrow = (size_t)bh * HD_;
    vt[(vrow + 2 * t)     * S_ + s] = (__bf16)vf[base + 2 * t];
    vt[(vrow + 2 * t + 1) * S_ + s] = (__bf16)vf[base + 2 * t + 1];
}

// ---------------------------------------------------------------------------
// Kernel 4: sliding-window flash attention. 1 wave per 16-query tile.
// Keys in (q-512, q]; processed in 32-key chunks with online softmax.
// Output bf16 scattered to [b, s, h*HD+hd] for the final projection GEMM.
// ---------------------------------------------------------------------------
__global__ __launch_bounds__(128) void attn_kernel(
    const __bf16* __restrict__ qm, const __bf16* __restrict__ km,
    const __bf16* __restrict__ vm, __bf16* __restrict__ om) {
    __shared__ __align__(16) __bf16 pshm[4][16][32];   // per-wave P transpose tile

    const int lane  = threadIdx.x & 31;
    const int wave  = threadIdx.x >> 5;
    const int lrow  = lane & 15;
    const int lhalf = (lane >> 4) << 3;

    int tile = blockIdx.x * 4 + wave;                 // 0..4095
    int b  = tile >> 11;
    int h  = (tile >> 7) & (H_ - 1);
    int s0 = (tile & 127) << 4;

    const size_t head  = ((size_t)(b * H_ + h)) * S_ * HD_;   // [s,hd] base
    const size_t headv = ((size_t)(b * H_ + h)) * HD_ * S_;   // [hd,s] base

    // Q as 4 A-operands (K = hd chunks of 32), held for the whole loop
    V16u qa[4];
#pragma unroll
    for (int kc = 0; kc < 4; ++kc)
        qa[kc] = load_op16(qm + head + (size_t)(s0 + lrow) * HD_ + kc * 32 + lhalf);

    v8f acc[8];
#pragma unroll
    for (int i = 0; i < 8; ++i) acc[i] = 0.0f;
    float mrun[8], lrun[8];
#pragma unroll
    for (int r = 0; r < 8; ++r) { mrun[r] = -1e30f; lrun[r] = 0.0f; }

    const float scale = 0.08838834764831845f;         // 1/sqrt(128)

    int lo = s0 - (WIN_ - 1);
    int j_start = (lo <= 0) ? 0 : (lo & ~31);
    for (int j0 = j_start; j0 <= s0 + 15; j0 += 32) {
        // ---- scores: Q @ K^T for 32 keys (2 N-tiles) ----
        v8f st[2];
        st[0] = 0.0f; st[1] = 0.0f;
#pragma unroll
        for (int nt = 0; nt < 2; ++nt) {
            const int key = j0 + nt * 16 + lrow;      // provably < S_
            const __bf16* krow = km + head + (size_t)key * HD_;
#pragma unroll
            for (int kc = 0; kc < 4; ++kc) {
                V16u bm = load_op16(krow + kc * 32 + lhalf);
                st[nt] = wmma_bf16(qa[kc].v, bm.v, st[nt]);
            }
        }

        // ---- mask + online softmax (rows live across 16-lane halves) ----
#pragma unroll
        for (int r = 0; r < 8; ++r) {
            int qrow = s0 + r + lhalf;
#pragma unroll
            for (int nt = 0; nt < 2; ++nt) {
                int key = j0 + nt * 16 + lrow;
                float sv = st[nt][r] * scale;
                bool ok = (key <= qrow) && (key > qrow - WIN_);
                st[nt][r] = ok ? sv : -1e30f;
            }
            float t = fmaxf(st[0][r], st[1][r]);
            t = fmaxf(t, __shfl_xor(t, 1, 32));
            t = fmaxf(t, __shfl_xor(t, 2, 32));
            t = fmaxf(t, __shfl_xor(t, 4, 32));
            t = fmaxf(t, __shfl_xor(t, 8, 32));
            float mnew  = fmaxf(mrun[r], t);
            float alpha = __expf(mrun[r] - mnew);
            mrun[r] = mnew;
            float p0 = __expf(st[0][r] - mnew);
            float p1 = __expf(st[1][r] - mnew);
            st[0][r] = p0; st[1][r] = p1;
            float rs = p0 + p1;
            rs += __shfl_xor(rs, 1, 32);
            rs += __shfl_xor(rs, 2, 32);
            rs += __shfl_xor(rs, 4, 32);
            rs += __shfl_xor(rs, 8, 32);
            lrun[r] = lrun[r] * alpha + rs;
#pragma unroll
            for (int nt2 = 0; nt2 < 8; ++nt2) acc[nt2][r] *= alpha;
        }

        // ---- transpose P (D layout -> A layout) through LDS ----
#pragma unroll
        for (int r = 0; r < 8; ++r) {
            int m = r + lhalf;
            pshm[wave][m][lrow]      = (__bf16)st[0][r];
            pshm[wave][m][16 + lrow] = (__bf16)st[1][r];
        }
        // same-wave LDS ops are in-order; compiler inserts s_wait_dscnt for VGPR use
        V16u pa;
        {
            const __bf16* p = &pshm[wave][lrow][lhalf];
            pa.h[0] = *(const v8bf*)(p);
            pa.h[1] = *(const v8bf*)(p + 16);
        }

        // ---- P @ V : V^T rows are key-contiguous -> clean 16B operand loads ----
#pragma unroll
        for (int nt2 = 0; nt2 < 8; ++nt2) {
            int hd = nt2 * 16 + lrow;
            const __bf16* vrow = vm + headv + (size_t)hd * S_ + j0 + lhalf;
            V16u bv;
            bv.h[0] = *(const v8bf*)(vrow);
            bv.h[1] = *(const v8bf*)(vrow + 16);
            acc[nt2] = wmma_bf16(pa.v, bv.v, acc[nt2]);
        }
    }

    // ---- normalize + store bf16 to [b, s, h*HD+hd] ----
#pragma unroll
    for (int r = 0; r < 8; ++r) {
        float rinv = 1.0f / lrun[r];
        int srow = s0 + r + lhalf;
#pragma unroll
        for (int nt2 = 0; nt2 < 8; ++nt2) {
            int hd = nt2 * 16 + lrow;
            om[((size_t)(b * S_ + srow)) * DIM_ + h * HD_ + hd] =
                (__bf16)(acc[nt2][r] * rinv);
        }
    }
}

// ---------------------------------------------------------------------------
// Host-side pipeline
// ---------------------------------------------------------------------------
extern "C" void kernel_launch(void* const* d_in, const int* in_sizes, int n_in,
                              void* d_out, int out_size, void* d_ws, size_t ws_size,
                              hipStream_t stream) {
    (void)in_sizes; (void)n_in; (void)out_size; (void)ws_size;
    const float* x  = (const float*)d_in[0];
    const float* Wq = (const float*)d_in[1];
    const float* Wk = (const float*)d_in[2];
    const float* Wv = (const float*)d_in[3];
    const float* Wo = (const float*)d_in[4];
    float* out = (float*)d_out;

    char* ws = (char*)d_ws;
    size_t off = 0;
    auto alloc = [&](size_t bytes) -> char* {
        char* p = ws + off;
        off += (bytes + 255) & ~(size_t)255;
        return p;
    };
    const size_t NX = (size_t)MTOT * DIM_;   // 8,388,608
    const size_t NW = (size_t)DIM_ * DIM_;   // 4,194,304

    __bf16* xb  = (__bf16*)alloc(NX * 2);
    __bf16* wqb = (__bf16*)alloc(NW * 2);
    __bf16* wkb = (__bf16*)alloc(NW * 2);
    __bf16* wvb = (__bf16*)alloc(NW * 2);
    __bf16* wob = (__bf16*)alloc(NW * 2);
    float*  qf  = (float*) alloc(NX * 4);
    float*  kf  = (float*) alloc(NX * 4);
    float*  vf  = (float*) alloc(NX * 4);
    __bf16* qbb = (__bf16*)alloc(NX * 2);
    __bf16* kbb = (__bf16*)alloc(NX * 2);
    __bf16* vtb = (__bf16*)alloc(NX * 2);
    __bf16* ab  = (__bf16*)alloc(NX * 2);

    // 1) stage inputs to bf16
    cvt_f32_bf16_v4<<<(int)(NX / 4 + 255) / 256, 256, 0, stream>>>(x,  xb,  (int)(NX / 4));
    cvt_f32_bf16_v4<<<(int)(NW / 4 + 255) / 256, 256, 0, stream>>>(Wq, wqb, (int)(NW / 4));
    cvt_f32_bf16_v4<<<(int)(NW / 4 + 255) / 256, 256, 0, stream>>>(Wk, wkb, (int)(NW / 4));
    cvt_f32_bf16_v4<<<(int)(NW / 4 + 255) / 256, 256, 0, stream>>>(Wv, wvb, (int)(NW / 4));
    cvt_f32_bf16_v4<<<(int)(NW / 4 + 255) / 256, 256, 0, stream>>>(Wo, wob, (int)(NW / 4));

    // 2) q/k/v projections (fused in one launch via grid.z), f32 out in [b,h,s,hd]
    gemm_nt_bf16<<<dim3(DIM_ / 128, MTOT / 128, 3), 256, 0, stream>>>(
        xb, wqb, wkb, wvb, qf, kf, vf, DIM_, DIM_, 0);

    // 3) RoPE + bf16 conversion; V transposed to [b,h,hd,s]
    rope_cvt<<<(B_ * H_ * S_ * 64) / 256, 256, 0, stream>>>(qf, kf, vf, qbb, kbb, vtb);

    // 4) sliding-window flash attention -> bf16 [b,s,dim]
    attn_kernel<<<(B_ * H_ * (S_ / 16)) / 4, 128, 0, stream>>>(qbb, kbb, vtb, ab);

    // 5) output projection -> f32 d_out
    gemm_nt_bf16<<<dim3(DIM_ / 128, MTOT / 128, 1), 256, 0, stream>>>(
        ab, wob, wob, wob, out, out, out, DIM_, DIM_, 1);
}
